// SelfAttentionFeatureExtractor_24919400252121
// MI455X (gfx1250) — matrix operand order repaired and checked by reading, not compiled, not run on hardware
//
#include <hip/hip_runtime.h>
#include <hip/hip_bf16.h>

// Problem constants (match reference)
#define BDIM 8
#define CDIM 4
#define BC   32          // B*C
#define TT   1024        // sequence length
#define DM   256         // d_in == d_model
#define E3   768         // 3*d_model

typedef __attribute__((ext_vector_type(16))) __bf16 bf16x16;
typedef __attribute__((ext_vector_type(8)))  __bf16 bf16x8;
typedef __attribute__((ext_vector_type(8)))  float  f32x8;
typedef __attribute__((ext_vector_type(4)))  unsigned int u32x4;
typedef __attribute__((ext_vector_type(8)))  int    i32x8;
typedef __attribute__((ext_vector_type(4)))  int    i32x4;

__device__ __forceinline__ f32x8 wmma_bf16(bf16x16 a, bf16x16 b, f32x8 c) {
    // D = A(16x32) * B(32x16) + C, f32 accumulate
    return __builtin_amdgcn_wmma_f32_16x16x32_bf16(false, a, false, b,
                                                   (short)0, c, false, false);
}

__device__ __forceinline__ bf16x16 ld_b16x16(const __bf16* p) {
    // two 16B loads (assume only 16B alignment)
    bf16x16 r;
    bf16x8 lo = *(const bf16x8*)p;
    bf16x8 hi = *(const bf16x8*)(p + 8);
#pragma unroll
    for (int i = 0; i < 8; ++i) { r[i] = lo[i]; r[8 + i] = hi[i]; }
    return r;
}

#if defined(__HIP_DEVICE_COMPILE__) && __has_builtin(__builtin_amdgcn_tensor_load_to_lds)
#define USE_TDM 1
// Issue a TDM 2D tile load: copies tile_d1 rows of tile_d0 elements (bf16)
// from global (row stride = stride0 elements) into contiguous LDS.
// D# bit layout per CDNA5 ISA ch.8 (group0 128b, group1 256b).
__device__ __forceinline__ void tdm_load_2d(unsigned lds_addr, const void* gaddr,
                                            unsigned tile_d0, unsigned tile_d1,
                                            unsigned stride0) {
    unsigned long long ga = (unsigned long long)(uintptr_t)gaddr;
    u32x4 g0;
    g0[0] = 1u;                                   // count=1, user mode, no gather
    g0[1] = lds_addr;                             // LDS byte address
    g0[2] = (unsigned)(ga & 0xffffffffu);         // global_addr[31:0]
    g0[3] = (unsigned)((ga >> 32) & 0x1ffffffu)   // global_addr[56:32]
          | (2u << 30);                           // type = 2 ("image")
    i32x8 g1;
    g1[0] = (int)(1u << 16);                      // data_size=1 (2B), wg_mask=0
    g1[1] = (int)((tile_d0 & 0xffffu) << 16);     // tensor_dim0[15:0]
    g1[2] = (int)((tile_d0 >> 16) |               // tensor_dim0[31:16]
                  ((tile_d1 & 0xffffu) << 16));   // tensor_dim1[15:0]
    g1[3] = (int)((tile_d1 >> 16) |               // tensor_dim1[31:16]
                  ((tile_d0 & 0xffffu) << 16));   // tile_dim0
    g1[4] = (int)(tile_d1 & 0xffffu);             // tile_dim1 (tile_dim2=0)
    g1[5] = (int)stride0;                         // tensor_dim0_stride[31:0]
    g1[6] = 0;                                    // stride0[47:32], stride1[15:0]
    g1[7] = 0;                                    // tensor_dim1_stride (unused 2D)
    i32x4 z4 = {0, 0, 0, 0};
#if __clang_major__ >= 23
    i32x8 z8 = {0, 0, 0, 0, 0, 0, 0, 0};
    __builtin_amdgcn_tensor_load_to_lds(g0, g1, z4, z4, z8, 0);
#else
    __builtin_amdgcn_tensor_load_to_lds(g0, g1, z4, z4, 0);
#endif
}
#else
#define USE_TDM 0
#endif

// ---------------------------------------------------------------------------
// Pass 1: QKV projection.  qkv[e,t] = sum_d w_qkv[e,d] * x[bc,d,t] + b_qkv[e]
// Each wave: one 16(e) x 64(t) tile, K-loop over d in steps of 32.
// Stores: Q -> (t,d) bf16, pre-scaled by 1/sqrt(256); K -> (t,d); V -> (d,t).
// ---------------------------------------------------------------------------
__global__ void __launch_bounds__(256)
qkv_proj_kernel(const float* __restrict__ x,
                const float* __restrict__ w_qkv,
                const float* __restrict__ b_qkv,
                __bf16* __restrict__ Q,
                __bf16* __restrict__ K,
                __bf16* __restrict__ V) {
    const int lane = threadIdx.x & 31;
    const int wv   = threadIdx.x >> 5;
    const int gid  = blockIdx.x * 8 + wv;          // 32*48*16 = 24576 jobs
    const int bc   = gid / (48 * 16);
    const int rem  = gid % (48 * 16);
    const int e0   = (rem >> 4) * 16;              // 0..752
    const int t0   = (rem & 15) * 64;              // 0..960
    const int half = lane >> 4;
    const int lm   = lane & 15;
    const int kbA  = half * 16;                    // A-operand K base
    const int kbB  = half * 16;                    // B-operand K base

    const float* xb = x + (size_t)bc * DM * TT;

    f32x8 acc[4] = {};
    for (int c = 0; c < 8; ++c) {
        const int d0 = c * 32;
        bf16x16 a;
        const float* ap = w_qkv + (size_t)(e0 + lm) * DM + d0 + kbA;
#pragma unroll
        for (int i = 0; i < 8; ++i) {
            a[i]     = (__bf16)ap[i];
            a[8 + i] = (__bf16)ap[8 + i];
        }
#pragma unroll
        for (int tt = 0; tt < 4; ++tt) {
            const int tcol = t0 + tt * 16 + lm;    // lane = column (t)
            bf16x16 b;
            const float* bp = xb + (size_t)(d0 + kbB) * TT + tcol;
#pragma unroll
            for (int i = 0; i < 16; ++i) b[i] = (__bf16)bp[(size_t)i * TT];
            acc[tt] = wmma_bf16(a, b, acc[tt]);
        }
    }

    const int seg = e0 >> 8;                       // 0=Q 1=K 2=V
#pragma unroll
    for (int r = 0; r < 8; ++r) {
        const int e = e0 + r + 8 * half;           // C-layout row
        const float bias = b_qkv[e];
#pragma unroll
        for (int tt = 0; tt < 4; ++tt) {
            const int t = t0 + tt * 16 + lm;       // C-layout col
            const float v = acc[tt][r] + bias;
            if (seg == 0) {
                Q[(size_t)bc * TT * DM + (size_t)t * DM + e] =
                    (__bf16)(v * 0.0625f);         // fold softmax scale
            } else if (seg == 1) {
                K[(size_t)bc * TT * DM + (size_t)t * DM + (e - 256)] = (__bf16)v;
            } else {
                V[(size_t)bc * DM * TT + (size_t)(e - 512) * TT + t] = (__bf16)v;
            }
        }
    }
}

// ---------------------------------------------------------------------------
// Pass 2: flash attention.  Block = 4 waves; K/V tiles (32 keys x 256 d)
// staged into double-buffered LDS via the Tensor Data Mover (wave 0 issues,
// TENSORcnt tracks, barrier publishes).  Each wave owns a 16-row Q tile and
// the full 16x256 f32 output accumulator.
// Dynamic LDS layout (bf16 elements):
//   sK: [2][32*256]  (32 KB)   [key][d]
//   sV: [2][256*32]  (32 KB)   [d][key]
//   sP: [4][512]     ( 4 KB)   per-wave P bounce
// ---------------------------------------------------------------------------
__global__ void __launch_bounds__(128)
flash_attn_kernel(const __bf16* __restrict__ Q,
                  const __bf16* __restrict__ K,
                  const __bf16* __restrict__ V,
                  const unsigned char* __restrict__ mask,
                  __bf16* __restrict__ AO) {
    extern __shared__ __bf16 smem[];
    __bf16* sK = smem;                 // 2 * 8192
    __bf16* sV = smem + 16384;         // 2 * 8192
    __bf16* sP = smem + 32768;         // 4 * 512

    const int tid  = threadIdx.x;
    const int lane = tid & 31;
    const int wv   = tid >> 5;
    const int bc   = blockIdx.x >> 4;
    const int qg   = blockIdx.x & 15;
    const int q0   = qg * 64 + wv * 16;
    const int half = lane >> 4;
    const int lm   = lane & 15;
    const int kbA  = half * 16;
    const int kbB  = half * 16;

    const __bf16* Qb = Q + (size_t)bc * TT * DM;
    const __bf16* Kb = K + (size_t)bc * TT * DM;
    const __bf16* Vb = V + (size_t)bc * DM * TT;
    const unsigned char* Mb = mask + (size_t)bc * TT * TT;

    // Q tile as 8 A-fragments (K = d, already scaled by 1/sqrt(d))
    bf16x16 aq[8];
#pragma unroll
    for (int c = 0; c < 8; ++c) {
        const __bf16* p = Qb + (size_t)(q0 + lm) * DM + c * 32 + kbA;
        bf16x8 lo = *(const bf16x8*)p;
        bf16x8 hi = *(const bf16x8*)(p + 8);
#pragma unroll
        for (int i = 0; i < 8; ++i) { aq[c][i] = lo[i]; aq[c][8 + i] = hi[i]; }
    }

    f32x8 o[16] = {};
    float mrow[8], lrow[8];
#pragma unroll
    for (int r = 0; r < 8; ++r) { mrow[r] = -3.0e38f; lrow[r] = 0.0f; }

    const int NK = TT / 32;

#if USE_TDM
    if (wv == 0) {   // prologue: DMA first K/V tile into buffer 0
        tdm_load_2d((unsigned)(uintptr_t)(void*)sK, Kb, DM, 32, DM);
        tdm_load_2d((unsigned)(uintptr_t)(void*)sV, Vb, 32, DM, TT);
    }
#endif

    for (int ks = 0; ks < NK; ++ks) {
        const int k0  = ks * 32;
        const int cur = ks & 1;
        __bf16* bufK = sK + cur * 8192;
        __bf16* bufV = sV + cur * 8192;

#if USE_TDM
        if (wv == 0) __builtin_amdgcn_s_wait_tensorcnt(0);  // tile `cur` landed
        __syncthreads();                                    // publish + reuse-safe
        if (wv == 0 && ks + 1 < NK) {                       // DMA next tile
            const int kn = k0 + 32;
            __bf16* nK = sK + (cur ^ 1) * 8192;
            __bf16* nV = sV + (cur ^ 1) * 8192;
            tdm_load_2d((unsigned)(uintptr_t)(void*)nK, Kb + (size_t)kn * DM,
                        DM, 32, DM);
            tdm_load_2d((unsigned)(uintptr_t)(void*)nV, Vb + kn,
                        32, DM, TT);
        }
#else
        __syncthreads();
        {   // cooperative staging fallback
            const uint4* src = (const uint4*)(Kb + (size_t)k0 * DM);
            uint4* dst = (uint4*)bufK;
#pragma unroll
            for (int i = 0; i < 8; ++i) dst[tid + 128 * i] = src[tid + 128 * i];
#pragma unroll
            for (int dd = 0; dd < 2; ++dd) {
                const int d = tid * 2 + dd;
                const uint4* vs = (const uint4*)(Vb + (size_t)d * TT + k0);
                uint4* vd = (uint4*)(bufV + d * 32);
#pragma unroll
                for (int i = 0; i < 4; ++i) vd[i] = vs[i];
            }
        }
        __syncthreads();
#endif

        // prefetch next mask tile while we compute (global_prefetch_b8)
        if (ks + 1 < NK)
            __builtin_prefetch(&Mb[(size_t)(q0 + lm) * TT + k0 + 32], 0, 1);

        // S = Q @ K^T : two 16x16 tiles (keys k0..+15, k0+16..+31)
        f32x8 s0 = {}, s1 = {};
#pragma unroll
        for (int c = 0; c < 8; ++c) {
            bf16x16 b0 = ld_b16x16(bufK + lm * DM + c * 32 + kbB);
            bf16x16 b1 = ld_b16x16(bufK + (16 + lm) * DM + c * 32 + kbB);
            s0 = wmma_bf16(aq[c], b0, s0);
            s1 = wmma_bf16(aq[c], b1, s1);
        }

        // Mask + online softmax (C-layout rows: row = r + 8*half)
        __bf16* pw = sP + wv * 512;
#pragma unroll
        for (int r = 0; r < 8; ++r) {
            const int q = q0 + r + 8 * half;
            const unsigned char m0 = Mb[(size_t)q * TT + k0 + lm];
            const unsigned char m1 = Mb[(size_t)q * TT + k0 + 16 + lm];
            const float v0 = m0 ? -__builtin_inff() : s0[r];
            const float v1 = m1 ? -__builtin_inff() : s1[r];
            float rowmax = fmaxf(v0, v1);
#pragma unroll
            for (int off = 8; off > 0; off >>= 1)
                rowmax = fmaxf(rowmax, __shfl_xor(rowmax, off, 32));
            const float mold  = mrow[r];
            const float mnew  = fmaxf(mold, rowmax);     // finite (mold finite)
            const float alpha = __expf(mold - mnew);
            const float p0 = __expf(v0 - mnew);
            const float p1 = __expf(v1 - mnew);
            float psum = p0 + p1;
#pragma unroll
            for (int off = 8; off > 0; off >>= 1)
                psum += __shfl_xor(psum, off, 32);
            lrow[r] = lrow[r] * alpha + psum;
            mrow[r] = mnew;
#pragma unroll
            for (int j = 0; j < 16; ++j) o[j][r] *= alpha;
            // bounce P through LDS: C-layout -> A-layout
            pw[(r + 8 * half) * 32 + lm]      = (__bf16)p0;
            pw[(r + 8 * half) * 32 + 16 + lm] = (__bf16)p1;
        }
        // same-wave DS ops are in-order; re-read P as an A operand
        bf16x16 apv;
        {
            const __bf16* pp = pw + lm * 32;
            bf16x8 lo = *(const bf16x8*)(pp + kbA);
            bf16x8 hi = *(const bf16x8*)(pp + 8 + kbA);
#pragma unroll
            for (int i = 0; i < 8; ++i) { apv[i] = lo[i]; apv[8 + i] = hi[i]; }
        }
        // O += P(16x32) @ V(32x256), 16 N-tiles
#pragma unroll
        for (int j = 0; j < 16; ++j) {
            bf16x16 bv = ld_b16x16(bufV + (j * 16 + lm) * 32 + kbB);
            o[j] = wmma_bf16(apv, bv, o[j]);
        }
    }

    // Finalize: divide by row sum, store (q,d) bf16
#pragma unroll
    for (int r = 0; r < 8; ++r) {
        const int q = q0 + r + 8 * half;
        const float inv = 1.0f / lrow[r];
#pragma unroll
        for (int j = 0; j < 16; ++j)
            AO[(size_t)bc * TT * DM + (size_t)q * DM + j * 16 + lm] =
                (__bf16)(o[j][r] * inv);
    }
}

// ---------------------------------------------------------------------------
// Pass 3: output projection.  out[t,e] = sum_d AO[t,d]*w_out[e,d] + b_out[e]
// Each wave: 16(q) x 64(e) tile.
// ---------------------------------------------------------------------------
__global__ void __launch_bounds__(256)
out_proj_kernel(const __bf16* __restrict__ AO,
                const float* __restrict__ w_out,
                const float* __restrict__ b_out,
                float* __restrict__ out) {
    const int lane = threadIdx.x & 31;
    const int wv   = threadIdx.x >> 5;
    const int gid  = blockIdx.x * 8 + wv;          // 32*64*4 = 8192 jobs
    const int bc   = gid >> 8;
    const int rem  = gid & 255;
    const int q0   = (rem >> 2) * 16;
    const int e0   = (rem & 3) * 64;
    const int half = lane >> 4;
    const int lm   = lane & 15;
    const int kbA  = half * 16;
    const int kbB  = half * 16;

    const __bf16* Ab = AO + (size_t)bc * TT * DM;

    f32x8 acc[4] = {};
    for (int c = 0; c < 8; ++c) {
        bf16x16 a;
        const __bf16* p = Ab + (size_t)(q0 + lm) * DM + c * 32 + kbA;
        bf16x8 lo = *(const bf16x8*)p;
        bf16x8 hi = *(const bf16x8*)(p + 8);
#pragma unroll
        for (int i = 0; i < 8; ++i) { a[i] = lo[i]; a[8 + i] = hi[i]; }
#pragma unroll
        for (int jj = 0; jj < 4; ++jj) {
            const int e = e0 + jj * 16 + lm;       // lane = column (e)
            const float* wp = w_out + (size_t)e * DM + c * 32 + kbB;
            bf16x16 b;
#pragma unroll
            for (int i = 0; i < 16; ++i) b[i] = (__bf16)wp[i];
            acc[jj] = wmma_bf16(a, b, acc[jj]);
        }
    }
#pragma unroll
    for (int jj = 0; jj < 4; ++jj) {
        const int e = e0 + jj * 16 + lm;
        const float bias = b_out[e];               // bias along N (lane) axis
#pragma unroll
        for (int r = 0; r < 8; ++r) {
            const int q = q0 + r + 8 * half;
            out[(size_t)bc * TT * DM + (size_t)q * DM + e] = acc[jj][r] + bias;
        }
    }
}

// ---------------------------------------------------------------------------
extern "C" void kernel_launch(void* const* d_in, const int* in_sizes, int n_in,
                              void* d_out, int out_size, void* d_ws, size_t ws_size,
                              hipStream_t stream) {
    const float*         x     = (const float*)d_in[0];
    const unsigned char* mask  = (const unsigned char*)d_in[1]; // jnp bool = 1B
    const float*         w_qkv = (const float*)d_in[2];
    const float*         b_qkv = (const float*)d_in[3];
    const float*         w_out = (const float*)d_in[4];
    const float*         b_out = (const float*)d_in[5];
    float*               out   = (float*)d_out;

    // Workspace: 4 x 16 MB bf16 buffers (Q, K, V, attn-out) = 64 MB
    __bf16* Q  = (__bf16*)d_ws;
    __bf16* K  = Q + (size_t)BC * TT * DM;
    __bf16* V  = K + (size_t)BC * TT * DM;
    __bf16* AO = V + (size_t)BC * TT * DM;

    const size_t smem2 = (size_t)(16384 + 16384 + 2048) * sizeof(__bf16); // 68 KB

    qkv_proj_kernel<<<3072, 256, 0, stream>>>(x, w_qkv, b_qkv, Q, K, V);
    flash_attn_kernel<<<512, 128, smem2, stream>>>(Q, K, V, mask, AO);
    out_proj_kernel<<<1024, 256, 0, stream>>>(AO, w_out, b_out, out);
}
